// TextGuidedMamba_46978352284126
// MI455X (gfx1250) — matrix operand order repaired and checked by reading, not compiled, not run on hardware
//
#include <hip/hip_runtime.h>
#include <hip/hip_bf16.h>

// ---------------- problem constants ----------------
#define B_SZ   8
#define DMODEL 96
#define NTOK   4096            // H*W
#define DSTATE 4
#define DI     192             // d_inner
#define DI2    384             // 2*d_inner
#define LSEQ   8192            // 2*NTOK
#define DTR    6               // dt_rank
#define EPSV   1e-5f
#define CH     128             // scan chunk length
#define NCH    (LSEQ / CH)     // 64 chunks per (b,d) chain

typedef __attribute__((ext_vector_type(16))) _Float16 v16h;
typedef __attribute__((ext_vector_type(8)))  float    v8f;
typedef __attribute__((ext_vector_type(4)))  int      v4i;

__device__ __forceinline__ float silu_f(float x) { return x / (1.f + expf(-x)); }

// ---------------- conversion / layout kernels ----------------

// NCHW fp32 -> (B*N, C) f16 token-major
__global__ void k_flatten_h(const float* __restrict__ src, _Float16* __restrict__ dst, int total)
{
    int i = blockIdx.x * blockDim.x + threadIdx.x;
    if (i >= total) return;
    int c = i % DMODEL;
    int t = (i / DMODEL) % NTOK;
    int b = i / (DMODEL * NTOK);
    dst[i] = (_Float16)src[((size_t)b * DMODEL + c) * NTOK + t];
}

__global__ void k_cvt_h(const float* __restrict__ s, _Float16* __restrict__ d, int n)
{
    int i = blockIdx.x * blockDim.x + threadIdx.x;
    if (i < n) d[i] = (_Float16)s[i];
}

// x_proj_w (14,192) -> zero-padded (16,192) f16
__global__ void k_xproj_pad(const float* __restrict__ s, _Float16* __restrict__ d)
{
    int i = blockIdx.x * blockDim.x + threadIdx.x;
    if (i >= 16 * DI) return;
    int r = i / DI, c = i % DI;
    d[i] = (r < (DTR + 2 * DSTATE)) ? (_Float16)s[r * DI + c] : (_Float16)0.f;
}

// ---------------- WMMA GEMM with LDS-staged weights ----------------
// C[m][n] = sum_k A[m][k] * Bw[n][k]; A:(M,K) f16 rm, Bw:(N,K) f16 rm, C fp32 (ldc).
// Block = 8 waves = 8 consecutive M-tiles, all sharing one N-tile (blockIdx.y).
// Bw tile (16 x K, K<=192 -> 6KB) staged in LDS once per block.
__global__ void k_wmma_gemm_nt(const _Float16* __restrict__ A,
                               const _Float16* __restrict__ Bw,
                               float* __restrict__ C,
                               int M, int N, int K, int ldc)
{
    __shared__ __align__(16) _Float16 bs[16 * 192];
    const int nt = blockIdx.y;

    // ---- stage the 16xK weight tile into LDS ----
    {
        const unsigned ndw = (unsigned)(16 * K) / 2;             // dwords
        const unsigned* src = (const unsigned*)(Bw + (size_t)nt * 16 * K);
        unsigned* dst = (unsigned*)bs;
#if __has_builtin(__builtin_amdgcn_global_load_async_to_lds_b128) && \
    __has_builtin(__builtin_amdgcn_s_wait_asynccnt)
        // gfx1250 async copy path (ASYNCcnt-tracked), 16B per op
        const unsigned n16 = ndw / 4;
        unsigned* srcnc = const_cast<unsigned*>(src);
        for (unsigned i = threadIdx.x; i < n16; i += blockDim.x)
            __builtin_amdgcn_global_load_async_to_lds_b128(
                (__attribute__((address_space(1))) v4i*)(srcnc + 4 * i),
                (__attribute__((address_space(3))) v4i*)(dst + 4 * i), 0, 0);
        __builtin_amdgcn_s_wait_asynccnt(0);
#else
        for (unsigned i = threadIdx.x; i < ndw; i += blockDim.x) dst[i] = src[i];
#endif
    }
    __syncthreads();

    const int wave = threadIdx.x >> 5;
    const int lane = threadIdx.x & 31;
    const int mt = blockIdx.x * 8 + wave;
    if (mt * 16 >= M) return;                 // wave-uniform: EXEC all-ones at WMMA

    const int ml = lane & 15;                 // A local row / B local col
    const int kh = lane >> 4;                 // K-half selector per ISA layout
    const _Float16* __restrict__ arow = A + (size_t)(mt * 16 + ml) * K;
    const _Float16* brow = bs + ml * K;       // LDS

    v8f acc = {};
    for (int k0 = 0; k0 < K; k0 += 32) {
        v16h af, bf;
        // A 16x32: VGPR v<4 -> K = k0 + 8*kh + 2v ; v>=4 -> K = k0 + 16 + 8*kh + 2(v-4)
#pragma unroll
        for (int v = 0; v < 4; ++v) {
            int ka = k0 + 8 * kh + 2 * v;
            af[2 * v]     = arow[ka];
            af[2 * v + 1] = arow[ka + 1];
            int kb = k0 + 16 + 8 * kh + 2 * v;
            af[8 + 2 * v]     = arow[kb];
            af[8 + 2 * v + 1] = arow[kb + 1];
        }
        // B 32x16: VGPR v<4 -> K = k0 + 8*kh + 2v ; v>=4 -> K = k0 + 16 + 8*kh + 2(v-4)
#pragma unroll
        for (int v = 0; v < 8; ++v) {
            int kb = k0 + ((v < 4) ? 8 * kh + 2 * v : 16 + 8 * kh + 2 * (v - 4));
            bf[2 * v]     = brow[kb];
            bf[2 * v + 1] = brow[kb + 1];
        }
        acc = __builtin_amdgcn_wmma_f32_16x16x32_f16(false, af, false, bf,
                                                     (short)0, acc, false, false);
    }
    // C/D layout: VGPR v: lanes 0-15 -> M=v, lanes 16-31 -> M=v+8; N = lane&15
#pragma unroll
    for (int v = 0; v < 8; ++v) {
        int row = mt * 16 + ((lane < 16) ? v : v + 8);
        C[(size_t)row * ldc + nt * 16 + ml] = acc[v];
    }
}

// ---------------- depthwise conv3x3 + BN(eval) + SiLU ----------------
__global__ void k_dwconv_bn_silu(const float* __restrict__ xz,
                                 const float* __restrict__ cw, const float* __restrict__ cb,
                                 const float* __restrict__ g,  const float* __restrict__ bb,
                                 const float* __restrict__ bm, const float* __restrict__ bv,
                                 float* __restrict__ xfor, _Float16* __restrict__ xforh,
                                 int tok_off)
{
    int i = blockIdx.x * blockDim.x + threadIdx.x;
    if (i >= B_SZ * NTOK * DI) return;
    int d = i % DI;
    int t = (i / DI) % NTOK;
    int b = i / (DI * NTOK);
    int h = t >> 6, w = t & 63;
    float acc = cb[d];
#pragma unroll
    for (int dh = -1; dh <= 1; ++dh)
#pragma unroll
        for (int dw = -1; dw <= 1; ++dw) {
            int hh = h + dh, ww = w + dw;
            if ((unsigned)hh < 64u && (unsigned)ww < 64u)
                acc += cw[d * 9 + (dh + 1) * 3 + (dw + 1)] *
                       xz[((size_t)b * NTOK + hh * 64 + ww) * DI2 + d];
        }
    acc = (acc - bm[d]) * rsqrtf(bv[d] + EPSV) * g[d] + bb[d];
    float v = silu_f(acc);
    size_t o = ((size_t)b * LSEQ + tok_off + t) * DI + d;
    xfor[o]  = v;
    xforh[o] = (_Float16)v;
}

// ---------------- dt projection + softplus ----------------
__global__ void k_delta(const float* __restrict__ xdbl, const float* __restrict__ dtw,
                        const float* __restrict__ dtb, float* __restrict__ delta)
{
    int i = blockIdx.x * blockDim.x + threadIdx.x;
    if (i >= B_SZ * LSEQ * DI) return;
    int d = i % DI;
    size_t tok = (size_t)i / DI;
    const float* r = xdbl + tok * 16;
    float a = dtb[d];
#pragma unroll
    for (int j = 0; j < DTR; ++j) a += r[j] * dtw[d * DTR + j];
    delta[i] = (a > 20.f) ? a : log1pf(expf(a));
}

// ================= chunked parallel selective scan =================
// h_l = exp(dl*A_n)*h_{l-1} + (dl*x_l)*B_n[l]  composed per chunk:
//   h_out = P*h_in + H,  P = exp(A_n * sum(dl)),  H = local scan from 0.

// phase 1: per-(b,d,chunk) local scan -> chunk aggregates (P, H)
__global__ void k_scan_local(const float* __restrict__ xfor, const float* __restrict__ delta,
                             const float* __restrict__ xdbl, const float* __restrict__ Alog,
                             float* __restrict__ cP, float* __restrict__ cH)
{
    int i = blockIdx.x * blockDim.x + threadIdx.x;
    if (i >= B_SZ * DI * NCH) return;
    int c = i % NCH;
    int d = (i / NCH) % DI;
    int b = i / (NCH * DI);
    float A0 = -expf(Alog[d * 4 + 0]);
    float A1 = -expf(Alog[d * 4 + 1]);
    float A2 = -expf(Alog[d * 4 + 2]);
    float A3 = -expf(Alog[d * 4 + 3]);
    size_t row0 = (size_t)b * LSEQ + (size_t)c * CH;
    const float* db = delta + row0 * DI + d;
    const float* xb = xfor  + row0 * DI + d;
    const float* bc = xdbl  + row0 * 16;
    float h0 = 0.f, h1 = 0.f, h2 = 0.f, h3 = 0.f, sd = 0.f;
    for (int l = 0; l < CH; ++l) {
        float dl = db[(size_t)l * DI];
        float xv = xb[(size_t)l * DI];
        const float* r = bc + l * 16;
        float dx = dl * xv;
        h0 = expf(dl * A0) * h0 + dx * r[6];
        h1 = expf(dl * A1) * h1 + dx * r[7];
        h2 = expf(dl * A2) * h2 + dx * r[8];
        h3 = expf(dl * A3) * h3 + dx * r[9];
        sd += dl;
        if (l + 1 < CH) {  // gfx1250 global_prefetch_b8
            __builtin_prefetch(db + (size_t)(l + 1) * DI, 0, 1);
            __builtin_prefetch(xb + (size_t)(l + 1) * DI, 0, 1);
        }
    }
    size_t o = (((size_t)b * DI + d) * NCH + c) * 4;
    cP[o + 0] = expf(A0 * sd); cP[o + 1] = expf(A1 * sd);
    cP[o + 2] = expf(A2 * sd); cP[o + 3] = expf(A3 * sd);
    cH[o + 0] = h0; cH[o + 1] = h1; cH[o + 2] = h2; cH[o + 3] = h3;
}

// phase 2: 64-step prefix over chunks per (b,d): hin[c] = incoming state of chunk c
__global__ void k_scan_prefix(const float* __restrict__ cP, const float* __restrict__ cH,
                              float* __restrict__ hin)
{
    int i = blockIdx.x * blockDim.x + threadIdx.x;
    if (i >= B_SZ * DI) return;
    size_t base = (size_t)i * NCH * 4;
    float h0 = 0.f, h1 = 0.f, h2 = 0.f, h3 = 0.f;
    for (int c = 0; c < NCH; ++c) {
        size_t o = base + (size_t)c * 4;
        hin[o + 0] = h0; hin[o + 1] = h1; hin[o + 2] = h2; hin[o + 3] = h3;
        h0 = cH[o + 0] + cP[o + 0] * h0;
        h1 = cH[o + 1] + cP[o + 1] * h1;
        h2 = cH[o + 2] + cP[o + 2] * h2;
        h3 = cH[o + 3] + cP[o + 3] * h3;
    }
}

// phase 3: re-run local scan with correct incoming state, emit y
__global__ void k_scan_apply(const float* __restrict__ xfor, const float* __restrict__ delta,
                             const float* __restrict__ xdbl, const float* __restrict__ Alog,
                             const float* __restrict__ Dsk, const float* __restrict__ hin,
                             float* __restrict__ y)
{
    int i = blockIdx.x * blockDim.x + threadIdx.x;
    if (i >= B_SZ * DI * NCH) return;
    int c = i % NCH;
    int d = (i / NCH) % DI;
    int b = i / (NCH * DI);
    float A0 = -expf(Alog[d * 4 + 0]);
    float A1 = -expf(Alog[d * 4 + 1]);
    float A2 = -expf(Alog[d * 4 + 2]);
    float A3 = -expf(Alog[d * 4 + 3]);
    float Dd = Dsk[d];
    size_t o = (((size_t)b * DI + d) * NCH + c) * 4;
    float h0 = hin[o + 0], h1 = hin[o + 1], h2 = hin[o + 2], h3 = hin[o + 3];
    size_t row0 = (size_t)b * LSEQ + (size_t)c * CH;
    const float* db = delta + row0 * DI + d;
    const float* xb = xfor  + row0 * DI + d;
    const float* bc = xdbl  + row0 * 16;
    float*       yb = y     + row0 * DI + d;
    for (int l = 0; l < CH; ++l) {
        float dl = db[(size_t)l * DI];
        float xv = xb[(size_t)l * DI];
        const float* r = bc + l * 16;
        float dx = dl * xv;
        h0 = expf(dl * A0) * h0 + dx * r[6];
        h1 = expf(dl * A1) * h1 + dx * r[7];
        h2 = expf(dl * A2) * h2 + dx * r[8];
        h3 = expf(dl * A3) * h3 + dx * r[9];
        yb[(size_t)l * DI] = h0 * r[10] + h1 * r[11] + h2 * r[12] + h3 * r[13] + Dd * xv;
    }
}

// ---------------- LayerNorm(d=192) + SiLU gate, emit f16 ----------------
__global__ void k_ln_gate(const float* __restrict__ y,
                          const float* __restrict__ xz_img, const float* __restrict__ xz_text,
                          const float* __restrict__ lg, const float* __restrict__ lb,
                          _Float16* __restrict__ yh)
{
    int tok = blockIdx.x;          // b*L + l
    int d   = threadIdx.x;         // 0..191
    int b = tok / LSEQ, l = tok % LSEQ;
    float v = y[(size_t)tok * DI + d];
    __shared__ float s1[DI], s2[DI], mv[2];
    s1[d] = v; s2[d] = v * v;
    __syncthreads();
    for (int s = 96; s >= 3; s >>= 1) {
        if (d < s) { s1[d] += s1[d + s]; s2[d] += s2[d + s]; }
        __syncthreads();
    }
    if (d == 0) {
        float su = s1[0] + s1[1] + s1[2];
        float sq = s2[0] + s2[1] + s2[2];
        float mu = su * (1.f / DI);
        mv[0] = mu;
        mv[1] = rsqrtf(sq * (1.f / DI) - mu * mu + EPSV);
    }
    __syncthreads();
    float nrm = (v - mv[0]) * mv[1] * lg[d] + lb[d];
    float zv = (l < NTOK)
        ? xz_img [((size_t)b * NTOK + l         ) * DI2 + DI + d]
        : xz_text[((size_t)b * NTOK + (l - NTOK)) * DI2 + DI + d];
    yh[(size_t)tok * DI + d] = (_Float16)(nrm * silu_f(zv));
}

// ---------------- interleave two streams per spatial token ----------------
__global__ void k_interleave(const float* __restrict__ outf, _Float16* __restrict__ fh)
{
    int i = blockIdx.x * blockDim.x + threadIdx.x;
    if (i >= B_SZ * NTOK * DI) return;
    int c = i % DMODEL;
    int s = (i / DMODEL) & 1;
    int t = (i / DI) % NTOK;
    int b = i / (DI * NTOK);
    fh[i] = (_Float16)outf[(((size_t)b * LSEQ) + s * NTOK + t) * DMODEL + c];
}

// ---------------- final transpose (B,N,C) -> (B,C,H,W) ----------------
__global__ void k_out_tr(const float* __restrict__ fin, float* __restrict__ out)
{
    int i = blockIdx.x * blockDim.x + threadIdx.x;
    if (i >= B_SZ * DMODEL * NTOK) return;
    int t = i % NTOK;
    int c = (i / NTOK) % DMODEL;
    int b = i / (NTOK * DMODEL);
    out[i] = fin[((size_t)b * NTOK + t) * DMODEL + c];
}

// ---------------- launcher ----------------
extern "C" void kernel_launch(void* const* d_in, const int* in_sizes, int n_in,
                              void* d_out, int out_size, void* d_ws, size_t ws_size,
                              hipStream_t stream)
{
    (void)in_sizes; (void)n_in; (void)out_size; (void)ws_size;

    const float* img       = (const float*)d_in[0];
    const float* text      = (const float*)d_in[1];
    const float* W_in_img  = (const float*)d_in[2];
    const float* W_in_text = (const float*)d_in[3];
    const float* cw_img    = (const float*)d_in[4];
    const float* cb_img    = (const float*)d_in[5];
    const float* bg_img    = (const float*)d_in[6];
    const float* bb_img    = (const float*)d_in[7];
    const float* bm_img    = (const float*)d_in[8];
    const float* bv_img    = (const float*)d_in[9];
    const float* cw_txt    = (const float*)d_in[10];
    const float* cb_txt    = (const float*)d_in[11];
    const float* bg_txt    = (const float*)d_in[12];
    const float* bb_txt    = (const float*)d_in[13];
    const float* bm_txt    = (const float*)d_in[14];
    const float* bv_txt    = (const float*)d_in[15];
    const float* x_proj_w  = (const float*)d_in[16];
    const float* dt_proj_w = (const float*)d_in[17];
    const float* dt_proj_b = (const float*)d_in[18];
    const float* A_log     = (const float*)d_in[19];
    const float* D_skip    = (const float*)d_in[20];
    const float* ln_g      = (const float*)d_in[21];
    const float* ln_b      = (const float*)d_in[22];
    const float* W_out     = (const float*)d_in[23];
    const float* W_fusion  = (const float*)d_in[24];
    float* out = (float*)d_out;

    // ---- workspace layout ----
    char* w = (char*)d_ws;
    size_t off = 0;
    auto alloc = [&](size_t bytes) -> void* {
        void* p = w + off;
        off += (bytes + 255) & ~(size_t)255;
        return p;
    };
    _Float16* img_h   = (_Float16*)alloc((size_t)B_SZ * NTOK * DMODEL * 2);
    _Float16* text_h  = (_Float16*)alloc((size_t)B_SZ * NTOK * DMODEL * 2);
    _Float16* Wii_h   = (_Float16*)alloc((size_t)DI2 * DMODEL * 2);
    _Float16* Wit_h   = (_Float16*)alloc((size_t)DI2 * DMODEL * 2);
    _Float16* Wxp_h   = (_Float16*)alloc((size_t)16 * DI * 2);
    _Float16* Wo_h    = (_Float16*)alloc((size_t)DMODEL * DI * 2);
    _Float16* Wf_h    = (_Float16*)alloc((size_t)DMODEL * DI * 2);
    float*    xz_img  = (float*)   alloc((size_t)B_SZ * NTOK * DI2 * 4);
    float*    xz_text = (float*)   alloc((size_t)B_SZ * NTOK * DI2 * 4);
    float*    x_for   = (float*)   alloc((size_t)B_SZ * LSEQ * DI * 4);
    _Float16* x_for_h = (_Float16*)alloc((size_t)B_SZ * LSEQ * DI * 2);   // reused as y_h
    float*    x_dbl   = (float*)   alloc((size_t)B_SZ * LSEQ * 16 * 4);
    float*    delta   = (float*)   alloc((size_t)B_SZ * LSEQ * DI * 4);   // reused as out_f
    float*    ybuf    = (float*)   alloc((size_t)B_SZ * LSEQ * DI * 4);
    float*    cP      = (float*)   alloc((size_t)B_SZ * DI * NCH * 4 * 4);
    float*    cH      = (float*)   alloc((size_t)B_SZ * DI * NCH * 4 * 4);
    float*    hin     = (float*)   alloc((size_t)B_SZ * DI * NCH * 4 * 4);
    _Float16* y_h     = x_for_h;                 // alias (x_for_h dead after x_dbl GEMM)
    float*    out_f   = delta;                   // alias (delta dead after scan)
    _Float16* fused_h = (_Float16*)xz_img;       // alias (xz dead after LN gate)
    float*    fin     = xz_text;                 // alias

    const int TB = 256;
    auto blocks = [](long n, int tb) { return (int)((n + tb - 1) / tb); };
    auto gemm_grid = [](int M, int N) { return dim3(M / 128, N / 16); };  // 8 waves/block

    // 1) flatten + convert activations / weights to f16
    k_flatten_h<<<blocks((long)B_SZ * NTOK * DMODEL, TB), TB, 0, stream>>>(img,  img_h,  B_SZ * NTOK * DMODEL);
    k_flatten_h<<<blocks((long)B_SZ * NTOK * DMODEL, TB), TB, 0, stream>>>(text, text_h, B_SZ * NTOK * DMODEL);
    k_cvt_h<<<blocks(DI2 * DMODEL, TB), TB, 0, stream>>>(W_in_img,  Wii_h, DI2 * DMODEL);
    k_cvt_h<<<blocks(DI2 * DMODEL, TB), TB, 0, stream>>>(W_in_text, Wit_h, DI2 * DMODEL);
    k_cvt_h<<<blocks(DMODEL * DI, TB), TB, 0, stream>>>(W_out,    Wo_h, DMODEL * DI);
    k_cvt_h<<<blocks(DMODEL * DI, TB), TB, 0, stream>>>(W_fusion, Wf_h, DMODEL * DI);
    k_xproj_pad<<<blocks(16 * DI, TB), TB, 0, stream>>>(x_proj_w, Wxp_h);

    // 2) in-projections: (32768,96) @ (384,96)^T -> (32768,384)
    k_wmma_gemm_nt<<<gemm_grid(B_SZ * NTOK, DI2), TB, 0, stream>>>(
        img_h,  Wii_h, xz_img,  B_SZ * NTOK, DI2, DMODEL, DI2);
    k_wmma_gemm_nt<<<gemm_grid(B_SZ * NTOK, DI2), TB, 0, stream>>>(
        text_h, Wit_h, xz_text, B_SZ * NTOK, DI2, DMODEL, DI2);

    // 3) depthwise conv + BN + SiLU -> x_for (fp32 + f16)
    k_dwconv_bn_silu<<<blocks((long)B_SZ * NTOK * DI, TB), TB, 0, stream>>>(
        xz_img,  cw_img, cb_img, bg_img, bb_img, bm_img, bv_img, x_for, x_for_h, 0);
    k_dwconv_bn_silu<<<blocks((long)B_SZ * NTOK * DI, TB), TB, 0, stream>>>(
        xz_text, cw_txt, cb_txt, bg_txt, bb_txt, bm_txt, bv_txt, x_for, x_for_h, NTOK);

    // 4) x_proj: (65536,192) @ (16,192)^T -> (65536,16)
    k_wmma_gemm_nt<<<gemm_grid(B_SZ * LSEQ, 16), TB, 0, stream>>>(
        x_for_h, Wxp_h, x_dbl, B_SZ * LSEQ, 16, DI, 16);

    // 5) delta = softplus(dt @ dt_proj_w^T + b)
    k_delta<<<blocks((long)B_SZ * LSEQ * DI, TB), TB, 0, stream>>>(
        x_dbl, dt_proj_w, dt_proj_b, delta);

    // 6) chunked parallel selective scan (3 phases)
    k_scan_local<<<blocks((long)B_SZ * DI * NCH, TB), TB, 0, stream>>>(
        x_for, delta, x_dbl, A_log, cP, cH);
    k_scan_prefix<<<blocks(B_SZ * DI, TB), TB, 0, stream>>>(cP, cH, hin);
    k_scan_apply<<<blocks((long)B_SZ * DI * NCH, TB), TB, 0, stream>>>(
        x_for, delta, x_dbl, A_log, D_skip, hin, ybuf);

    // 7) LayerNorm + SiLU(z) gate -> f16
    k_ln_gate<<<B_SZ * LSEQ, DI, 0, stream>>>(ybuf, xz_img, xz_text, ln_g, ln_b, y_h);

    // 8) out-proj: (65536,192) @ (96,192)^T -> (65536,96)
    k_wmma_gemm_nt<<<gemm_grid(B_SZ * LSEQ, DMODEL), TB, 0, stream>>>(
        y_h, Wo_h, out_f, B_SZ * LSEQ, DMODEL, DI, DMODEL);

    // 9) interleave streams -> (32768,192) f16
    k_interleave<<<blocks((long)B_SZ * NTOK * DI, TB), TB, 0, stream>>>(out_f, fused_h);

    // 10) fusion: (32768,192) @ (96,192)^T -> (32768,96)
    k_wmma_gemm_nt<<<gemm_grid(B_SZ * NTOK, DMODEL), TB, 0, stream>>>(
        fused_h, Wf_h, fin, B_SZ * NTOK, DMODEL, DI, DMODEL);

    // 11) transpose to (B,C,H,W)
    k_out_tr<<<blocks((long)B_SZ * DMODEL * NTOK, TB), TB, 0, stream>>>(fin, out);
}
